// MultiheadLatentAttention_63273458204778
// MI455X (gfx1250) — compile-verified, matrix operand order checked by reading, not run on hardware
//
#include <hip/hip_runtime.h>
#include <cstdint>
#include <cstddef>

typedef __attribute__((ext_vector_type(16))) __bf16 v16bf;
typedef __attribute__((ext_vector_type(8)))  float  v8f;

#define B_    2
#define T_    2048
#define HID_  2048
#define H_    16
#define DQ_   128
#define DV_   128
#define R_    32
#define DCQ_  768
#define DCKV_ 512
#define DCO_  512
#define BT_   (B_ * T_)
#define SCALE_ 0.08838834764831845f
#define EPS_  1e-6f

static __device__ __forceinline__ __bf16 f2bf(float f) {
  union { float f; uint32_t u; } a; a.f = f;
  uint32_t u = a.u;
  uint32_t r = u + 0x7FFFu + ((u >> 16) & 1u);   // round-to-nearest-even
  unsigned short h = (unsigned short)(r >> 16);
  return __builtin_bit_cast(__bf16, h);
}

// Generic pointers into LDS carry the wave-relative LDS offset in their low
// 32 bits (ISA 10.2: LDS aperture -> low 32 bits are the LDS offset).
static __device__ __forceinline__ uint32_t lds_off(const void* p) {
  return (uint32_t)(uintptr_t)p;
}

// gfx1250 async global->LDS copy, 16 bytes per lane, tracked by ASYNCcnt.
// Data bypasses the register file entirely.
static __device__ __forceinline__ void async_copy_b128(uint32_t lds_addr,
                                                       const void* gptr) {
  unsigned long long ga = (unsigned long long)(uintptr_t)gptr;
  asm volatile("global_load_async_to_lds_b128 %0, %1, off"
               :: "v"(lds_addr), "v"(ga)
               : "memory");
}
static __device__ __forceinline__ void wait_async0() {
  asm volatile("s_wait_asynccnt 0" ::: "memory");
}

// Load one 16-lane-row x 32-K bf16 WMMA operand fragment.
// Per CDNA5 ISA 16-bit A/B layout: lanes 0-15 hold K = [0..7] and [16..23],
// lanes 16-31 hold K = [8..15] and [24..31] -> two contiguous 16B chunks.
static __device__ __forceinline__ v16bf load_frag_row(const __bf16* rowbase, int khalf) {
  union { v16bf v; uint4 u[2]; } t;
  t.u[0] = *(const uint4*)(rowbase + khalf);
  t.u[1] = *(const uint4*)(rowbase + khalf + 16);
  return t.v;
}

// ---------------------------------------------------------------------------
// f32 -> bf16 convert (grid-stride)
// ---------------------------------------------------------------------------
__global__ void cvt_f32_bf16_kernel(const float* __restrict__ in,
                                    __bf16* __restrict__ out, size_t n) {
  size_t i = (size_t)blockIdx.x * blockDim.x + threadIdx.x;
  size_t stride = (size_t)gridDim.x * blockDim.x;
  for (; i < n; i += stride) out[i] = f2bf(in[i]);
}

// ---------------------------------------------------------------------------
// Generic bf16 WMMA GEMM: C[M,N] (f32) = A[M,K] (bf16, row-major)
//                                       @ W[N,K]^T (bf16, row-major weights)
// 128x128 block tile, K-step 32, 8 waves each computing 64x32.
// Double-buffered LDS filled with async-to-LDS copies so the next tile
// streams in while the current tile's WMMAs execute.
// ---------------------------------------------------------------------------
__global__ __launch_bounds__(256) void gemm_bf16_kernel(
    const __bf16* __restrict__ A, const __bf16* __restrict__ W,
    float* __restrict__ C, int M, int N, int K) {
  __shared__ alignas(16) __bf16 sA[2][128 * 32];
  __shared__ alignas(16) __bf16 sB[2][128 * 32];

  const int tiles_n = N >> 7;
  const int bm = blockIdx.x / tiles_n;
  const int bn = blockIdx.x % tiles_n;
  const int tid  = threadIdx.x;
  const int wave = tid >> 5;
  const int lane = tid & 31;
  const int lrow = lane & 15;
  const int khalf = (lane >> 4) << 3;
  const int vhalf = (lane >> 4) << 3;
  const int wrow = (wave & 1) * 64;   // 2 wave-rows of 64
  const int wcol = (wave >> 1) * 32;  // 4 wave-cols of 32

  // This thread's two 16B chunks per tile (512 uint4 per 128x32 tile pair)
  const int r0 = tid >> 2,          s0 = (tid & 3) * 8;
  const int r1 = (tid + 256) >> 2,  s1 = ((tid + 256) & 3) * 8;
  const __bf16* Abase = A + (size_t)bm * 128 * K;
  const __bf16* Wbase = W + (size_t)bn * 128 * K;

  v8f acc[4][2];
#pragma unroll
  for (int i = 0; i < 4; ++i)
#pragma unroll
    for (int j = 0; j < 2; ++j) acc[i][j] = (v8f){0,0,0,0,0,0,0,0};

  auto issue_tile = [&](int buf, int k0) {
    async_copy_b128(lds_off(&sA[buf][r0 * 32 + s0]), Abase + (size_t)r0 * K + k0 + s0);
    async_copy_b128(lds_off(&sA[buf][r1 * 32 + s1]), Abase + (size_t)r1 * K + k0 + s1);
    async_copy_b128(lds_off(&sB[buf][r0 * 32 + s0]), Wbase + (size_t)r0 * K + k0 + s0);
    async_copy_b128(lds_off(&sB[buf][r1 * 32 + s1]), Wbase + (size_t)r1 * K + k0 + s1);
  };

  const int nk = K >> 5;
  issue_tile(0, 0);
  wait_async0();
  __syncthreads();

  for (int kt = 0; kt < nk; ++kt) {
    const int buf = kt & 1;
    if (kt + 1 < nk) issue_tile(buf ^ 1, (kt + 1) * 32);   // overlap with WMMA

    v16bf af[4], bfz[2];
#pragma unroll
    for (int i = 0; i < 4; ++i)
      af[i] = load_frag_row(&sA[buf][(wrow + i * 16 + lrow) * 32], khalf);
#pragma unroll
    for (int j = 0; j < 2; ++j)
      bfz[j] = load_frag_row(&sB[buf][(wcol + j * 16 + lrow) * 32], khalf);

#pragma unroll
    for (int i = 0; i < 4; ++i)
#pragma unroll
      for (int j = 0; j < 2; ++j)
        acc[i][j] = __builtin_amdgcn_wmma_f32_16x16x32_bf16(
            false, af[i], false, bfz[j], (short)0, acc[i][j], false, false);

    wait_async0();       // my async fills done
    __syncthreads();     // everyone done reading buf / writing buf^1
  }

  // Epilogue: one base pointer, constant relative offsets.
  float* cp0 = C + (size_t)(bm * 128 + wrow + vhalf) * N + (bn * 128 + wcol + lrow);
#pragma unroll
  for (int i = 0; i < 4; ++i)
#pragma unroll
    for (int j = 0; j < 2; ++j)
#pragma unroll
      for (int v = 0; v < 8; ++v)
        cp0[(size_t)(i * 16 + v) * N + j * 16] = acc[i][j][v];
}

// ---------------------------------------------------------------------------
// RMS norm: one block per row; f32 in, bf16 out
// ---------------------------------------------------------------------------
__global__ __launch_bounds__(256) void rms_bf16_kernel(
    const float* __restrict__ in, const float* __restrict__ w,
    __bf16* __restrict__ out, int N) {
  const int row = blockIdx.x;
  const float* x = in + (size_t)row * N;
  float s = 0.f;
  for (int i = threadIdx.x; i < N; i += 256) { float v = x[i]; s += v * v; }
  __shared__ float red[256];
  red[threadIdx.x] = s;
  __syncthreads();
  for (int off = 128; off > 0; off >>= 1) {
    if (threadIdx.x < off) red[threadIdx.x] += red[threadIdx.x + off];
    __syncthreads();
  }
  const float inv = rsqrtf(red[0] / (float)N + EPS_);
  for (int i = threadIdx.x; i < N; i += 256)
    out[(size_t)row * N + i] = f2bf(w[i] * x[i] * inv);
}

// ---------------------------------------------------------------------------
// RoPE + SiLU gating prep.  Writes Q,K head-major [b][h][t][128] bf16 and
// V transposed [b][h][d][t] bf16 (so attention's PV B-operand loads are
// contiguous).  One block per (b,t).
// ---------------------------------------------------------------------------
__global__ __launch_bounds__(256) void qkv_prep_kernel(
    const float* __restrict__ qf,   // [BT, 2048]
    const float* __restrict__ kvf,  // [BT, 4096]  (k_flat | v_flat)
    const float* __restrict__ gf,   // [BT, 2048]
    const float* __restrict__ cosT, // [T, 32]
    const float* __restrict__ sinT, // [T, 32]
    __bf16* __restrict__ Qh, __bf16* __restrict__ Kh,
    __bf16* __restrict__ Vt) {
  const int bt = blockIdx.x;
  const int b = bt / T_;
  const int t = bt % T_;
  for (int i = threadIdx.x; i < H_ * DQ_; i += 256) {
    const int h = i >> 7;
    const int d = i & 127;
    float qv = qf[(size_t)bt * (H_ * DQ_) + i];
    float kv = kvf[(size_t)bt * (H_ * (DQ_ + DV_)) + i];
    float vv = kvf[(size_t)bt * (H_ * (DQ_ + DV_)) + H_ * DQ_ + i];
    float g  = gf[(size_t)bt * (H_ * DV_) + i];
    g = g / (1.f + __expf(-g));   // silu
    vv *= g;
    if (d < R_) {
      const float c = cosT[t * R_ + d];
      const float s = sinT[t * R_ + d];
      const int dp = (d < R_ / 2) ? d + R_ / 2 : d - R_ / 2;
      const float qp = qf[(size_t)bt * (H_ * DQ_) + h * 128 + dp];
      const float kp = kvf[(size_t)bt * (H_ * (DQ_ + DV_)) + h * 128 + dp];
      const float rq = (d < R_ / 2) ? -qp : qp;   // rotate_half
      const float rk = (d < R_ / 2) ? -kp : kp;
      qv = qv * c + rq * s;
      kv = kv * c + rk * s;
    }
    const size_t o = ((size_t)(b * H_ + h) * T_ + t) * 128 + d;
    Qh[o] = f2bf(qv);
    Kh[o] = f2bf(kv);
    Vt[((size_t)(b * H_ + h) * 128 + d) * T_ + t] = f2bf(vv);
  }
}

// ---------------------------------------------------------------------------
// Flash attention (causal).  Grid = B*H*(T/128) blocks, 256 threads = 8 waves.
// Each wave owns 16 query rows; Q fragments in registers; K tile [key][d] and
// V^T tile [d][key] filled via async-to-LDS; per-wave P staging in LDS for
// the PV WMMA.
// ---------------------------------------------------------------------------
__global__ __launch_bounds__(256) void attn_kernel(
    const __bf16* __restrict__ Qh, const __bf16* __restrict__ Kh,
    const __bf16* __restrict__ Vt, __bf16* __restrict__ Out) {
  constexpr int NQB = T_ / 128;
  const int bh = blockIdx.x / NQB;
  const int qb = blockIdx.x % NQB;
  const int b = bh / H_;
  const int h = bh % H_;

  __shared__ alignas(16) __bf16 sK[128 * 128];
  __shared__ alignas(16) __bf16 sV[128 * 128];
  __shared__ alignas(16) __bf16 sP[8 * 16 * 128];

  const int tid  = threadIdx.x;
  const int wave = tid >> 5;
  const int lane = tid & 31;
  const int lrow = lane & 15;
  const int khalf = (lane >> 4) << 3;
  const int vhalf = (lane >> 4) << 3;

  // Q A-fragments for this wave's 16 rows (K-dim = 128 -> 4 fragments)
  v16bf qfrag[4];
  {
    const __bf16* qp =
        Qh + ((size_t)bh * T_ + (size_t)qb * 128 + wave * 16 + lrow) * 128;
#pragma unroll
    for (int ks = 0; ks < 4; ++ks) {
      union { v16bf v; uint4 u[2]; } t;
      t.u[0] = *(const uint4*)(qp + ks * 32 + khalf);
      t.u[1] = *(const uint4*)(qp + ks * 32 + khalf + 16);
      qfrag[ks] = t.v;
    }
  }

  v8f oacc[8];
#pragma unroll
  for (int i = 0; i < 8; ++i) oacc[i] = (v8f){0,0,0,0,0,0,0,0};
  float rowm[8], rowl[8];
#pragma unroll
  for (int v = 0; v < 8; ++v) { rowm[v] = -1e30f; rowl[v] = 0.f; }

  __bf16* myP = sP + wave * (16 * 128);
  const __bf16* Kbase = Kh + (size_t)bh * T_ * 128;
  const __bf16* Vbase = Vt + (size_t)bh * 128 * T_;
  const int r8 = tid >> 4, s8 = (tid & 15) * 8;   // 2048 uint4 per 128x128 tile

  for (int jb = 0; jb <= qb; ++jb) {
    // async fill of K tile [key][d] and V^T tile [d][key]
#pragma unroll
    for (int i = 0; i < 8; ++i) {
      const int r = r8 + i * 16;
      async_copy_b128(lds_off(sK + r * 128 + s8),
                      Kbase + ((size_t)jb * 128 + r) * 128 + s8);
      async_copy_b128(lds_off(sV + r * 128 + s8),
                      Vbase + (size_t)r * T_ + jb * 128 + s8);
    }
    wait_async0();
    __syncthreads();

    // S = Q K^T  (16 x 128 per wave)
    v8f sacc[8];
#pragma unroll
    for (int nt = 0; nt < 8; ++nt) {
      v8f a = (v8f){0,0,0,0,0,0,0,0};
#pragma unroll
      for (int ks = 0; ks < 4; ++ks) {
        v16bf bfrag = load_frag_row(sK + (nt * 16 + lrow) * 128 + ks * 32, khalf);
        a = __builtin_amdgcn_wmma_f32_16x16x32_bf16(false, qfrag[ks], false, bfrag,
                                                    (short)0, a, false, false);
      }
      sacc[nt] = a;
    }

    // scale + causal mask (diagonal block only)
#pragma unroll
    for (int nt = 0; nt < 8; ++nt)
#pragma unroll
      for (int v = 0; v < 8; ++v) {
        float s = sacc[nt][v] * SCALE_;
        if (jb == qb) {
          int n = nt * 16 + lrow;
          int m = wave * 16 + vhalf + v;
          if (n > m) s = -1e30f;
        }
        sacc[nt][v] = s;
      }

    // online softmax: row max -> alpha -> P -> row sum
    float rm[8];
#pragma unroll
    for (int v = 0; v < 8; ++v) rm[v] = sacc[0][v];
#pragma unroll
    for (int nt = 1; nt < 8; ++nt)
#pragma unroll
      for (int v = 0; v < 8; ++v) rm[v] = fmaxf(rm[v], sacc[nt][v]);
#pragma unroll
    for (int v = 0; v < 8; ++v) {
      float x = rm[v];
      for (int off = 1; off < 16; off <<= 1) x = fmaxf(x, __shfl_xor(x, off, 32));
      rm[v] = x;
    }
    float alpha[8], rsum[8];
#pragma unroll
    for (int v = 0; v < 8; ++v) {
      float mnew = fmaxf(rowm[v], rm[v]);
      alpha[v] = __expf(rowm[v] - mnew);
      rowm[v] = mnew;
      rsum[v] = 0.f;
    }
#pragma unroll
    for (int nt = 0; nt < 8; ++nt)
#pragma unroll
      for (int v = 0; v < 8; ++v) {
        float pe = __expf(sacc[nt][v] - rowm[v]);
        rsum[v] += pe;
        myP[(vhalf + v) * 128 + nt * 16 + lrow] = f2bf(pe);
      }
#pragma unroll
    for (int v = 0; v < 8; ++v) {
      float x = rsum[v];
      for (int off = 1; off < 16; off <<= 1) x += __shfl_xor(x, off, 32);
      rowl[v] = rowl[v] * alpha[v] + x;
    }
#pragma unroll
    for (int dt = 0; dt < 8; ++dt)
#pragma unroll
      for (int v = 0; v < 8; ++v) oacc[dt][v] *= alpha[v];

    // O += P * V   (P: 16x128 A-operand, V^T tile gives contiguous B-operand)
#pragma unroll
    for (int ks = 0; ks < 4; ++ks) {
      v16bf pfrag = load_frag_row(myP + lrow * 128 + ks * 32, khalf);
#pragma unroll
      for (int dt = 0; dt < 8; ++dt) {
        v16bf vfrag = load_frag_row(sV + (dt * 16 + lrow) * 128 + ks * 32, khalf);
        oacc[dt] = __builtin_amdgcn_wmma_f32_16x16x32_bf16(false, pfrag, false, vfrag,
                                                           (short)0, oacc[dt], false, false);
      }
    }
    __syncthreads();
  }

  // normalize + store to [B*T, H*DV] bf16
  __bf16* op0 = Out + ((size_t)b * T_ + qb * 128 + wave * 16 + vhalf) * (H_ * DV_)
                    + (h * 128 + lrow);
#pragma unroll
  for (int dt = 0; dt < 8; ++dt)
#pragma unroll
    for (int v = 0; v < 8; ++v)
      op0[(size_t)v * (H_ * DV_) + dt * 16] = f2bf(oacc[dt][v] / rowl[v]);
}

// ---------------------------------------------------------------------------
// Host orchestration
// ---------------------------------------------------------------------------
extern "C" void kernel_launch(void* const* d_in, const int* in_sizes, int n_in,
                              void* d_out, int out_size, void* d_ws, size_t ws_size,
                              hipStream_t stream) {
  (void)in_sizes; (void)n_in; (void)out_size; (void)ws_size;
  const float* x    = (const float*)d_in[0];
  const float* cosT = (const float*)d_in[1];
  const float* sinT = (const float*)d_in[2];
  const float* Wqs  = (const float*)d_in[3];
  const float* qsn  = (const float*)d_in[4];
  const float* Wkvs = (const float*)d_in[5];
  const float* kvsn = (const float*)d_in[6];
  const float* Wqp  = (const float*)d_in[7];
  const float* Wkvp = (const float*)d_in[8];
  const float* Wg   = (const float*)d_in[9];
  const float* Wop  = (const float*)d_in[10];
  const float* opn  = (const float*)d_in[11];
  const float* Wos  = (const float*)d_in[12];
  float* out = (float*)d_out;

  char* base = (char*)d_ws;
  size_t off = 0;
  auto alloc = [&](size_t elems, size_t esz) -> void* {
    void* r = base + off;
    off += (elems * esz + 255) & ~(size_t)255;
    return r;
  };
  __bf16* xb     = (__bf16*)alloc((size_t)BT_ * HID_, 2);
  __bf16* wqs_b  = (__bf16*)alloc((size_t)DCQ_ * HID_, 2);
  __bf16* wkvs_b = (__bf16*)alloc((size_t)DCKV_ * HID_, 2);
  __bf16* wqp_b  = (__bf16*)alloc((size_t)(H_ * DQ_) * DCQ_, 2);
  __bf16* wkvp_b = (__bf16*)alloc((size_t)(H_ * (DQ_ + DV_)) * DCKV_, 2);
  __bf16* wg_b   = (__bf16*)alloc((size_t)(H_ * DV_) * HID_, 2);
  __bf16* wop_b  = (__bf16*)alloc((size_t)DCO_ * (H_ * DV_), 2);
  __bf16* wos_b  = (__bf16*)alloc((size_t)HID_ * DCO_, 2);
  __bf16* qs_b   = (__bf16*)alloc((size_t)BT_ * DCQ_, 2);
  __bf16* kvs_b  = (__bf16*)alloc((size_t)BT_ * DCKV_, 2);
  __bf16* Qh     = (__bf16*)alloc((size_t)B_ * H_ * T_ * DQ_, 2);
  __bf16* Kh     = (__bf16*)alloc((size_t)B_ * H_ * T_ * DQ_, 2);
  __bf16* Vt     = (__bf16*)alloc((size_t)B_ * H_ * DV_ * T_, 2);
  __bf16* attn_b = (__bf16*)alloc((size_t)BT_ * (H_ * DV_), 2);
  __bf16* o_b    = (__bf16*)alloc((size_t)BT_ * DCO_, 2);
  float*  qs_f   = (float*)alloc((size_t)BT_ * DCQ_, 4);
  float*  kvs_f  = (float*)alloc((size_t)BT_ * DCKV_, 4);
  float*  q_f    = (float*)alloc((size_t)BT_ * (H_ * DQ_), 4);
  float*  kv_f   = (float*)alloc((size_t)BT_ * (H_ * (DQ_ + DV_)), 4);
  float*  g_f    = (float*)alloc((size_t)BT_ * (H_ * DV_), 4);
  float*  o_f    = (float*)alloc((size_t)BT_ * DCO_, 4);

  auto cvt = [&](const float* in, __bf16* o, size_t n) {
    cvt_f32_bf16_kernel<<<1024, 256, 0, stream>>>(in, o, n);
  };
  auto gemm = [&](const __bf16* A, const __bf16* W, float* C, int M, int N, int K) {
    gemm_bf16_kernel<<<(M / 128) * (N / 128), 256, 0, stream>>>(A, W, C, M, N, K);
  };

  // bf16 conversions
  cvt(x,    xb,     (size_t)BT_ * HID_);
  cvt(Wqs,  wqs_b,  (size_t)DCQ_ * HID_);
  cvt(Wkvs, wkvs_b, (size_t)DCKV_ * HID_);
  cvt(Wqp,  wqp_b,  (size_t)(H_ * DQ_) * DCQ_);
  cvt(Wkvp, wkvp_b, (size_t)(H_ * (DQ_ + DV_)) * DCKV_);
  cvt(Wg,   wg_b,   (size_t)(H_ * DV_) * HID_);
  cvt(Wop,  wop_b,  (size_t)DCO_ * (H_ * DV_));
  cvt(Wos,  wos_b,  (size_t)HID_ * DCO_);

  // shared projections + RMS
  gemm(xb, wqs_b, qs_f, BT_, DCQ_, HID_);
  rms_bf16_kernel<<<BT_, 256, 0, stream>>>(qs_f, qsn, qs_b, DCQ_);
  gemm(xb, wkvs_b, kvs_f, BT_, DCKV_, HID_);
  rms_bf16_kernel<<<BT_, 256, 0, stream>>>(kvs_f, kvsn, kvs_b, DCKV_);

  // private projections + gates
  gemm(qs_b,  wqp_b,  q_f,  BT_, H_ * DQ_, DCQ_);
  gemm(kvs_b, wkvp_b, kv_f, BT_, H_ * (DQ_ + DV_), DCKV_);
  gemm(xb,    wg_b,   g_f,  BT_, H_ * DV_, HID_);

  // RoPE + SiLU gating, head-major layouts (V transposed)
  qkv_prep_kernel<<<BT_, 256, 0, stream>>>(q_f, kv_f, g_f, cosT, sinT, Qh, Kh, Vt);

  // causal flash attention
  attn_kernel<<<B_ * H_ * (T_ / 128), 256, 0, stream>>>(Qh, Kh, Vt, attn_b);

  // output projections
  gemm(attn_b, wop_b, o_f, BT_, DCO_, H_ * DV_);
  rms_bf16_kernel<<<BT_, 256, 0, stream>>>(o_f, opn, o_b, DCO_);
  gemm(o_b, wos_b, out, BT_, HID_, DCO_);
}
